// SubScheduler_55473797595590
// MI455X (gfx1250) — matrix operand order repaired and checked by reading, not compiled, not run on hardware
//
#include <hip/hip_runtime.h>
#include <math.h>

typedef _Float16 v16h __attribute__((ext_vector_type(16)));
typedef float    v8f  __attribute__((ext_vector_type(8)));
typedef float    f32x4 __attribute__((ext_vector_type(4)));
typedef float    f32x2 __attribute__((ext_vector_type(2)));
typedef int      i32x4 __attribute__((ext_vector_type(4)));
typedef int      i32x2 __attribute__((ext_vector_type(2)));

#define N_PTS 1024
#define D_IN  64
#define H1    128
#define H2    64
#define H1S   136   // padded LDS row stride (halves): start banks 4m apart -> conflict-free b128
#define H2S   68    // padded LDS row stride (f32) for 64-f32 rows
#define EPSG  1e-10f

// ---------------------------------------------------------------------------
// Kernel 1: Apre[i] = X[i] @ W1[:64,:] + b1 ; Bpre[i] = X[i] @ W1[64:,:]
// ---------------------------------------------------------------------------
__global__ void precompute_AB(const float* __restrict__ X,
                              const float* __restrict__ W1,
                              const float* __restrict__ b1,
                              float* __restrict__ Apre,
                              float* __restrict__ Bpre) {
  const int i = blockIdx.x;    // point index 0..1023
  const int t = threadIdx.x;   // output column 0..127
  __shared__ float sx[D_IN];
  if (t < D_IN) sx[t] = X[i * D_IN + t];
  __syncthreads();
  float a = b1[t];   // fold b1 into Apre (added once per pair)
  float b = 0.f;
#pragma unroll 8
  for (int k = 0; k < D_IN; ++k) {
    float xv = sx[k];
    a += xv * W1[k * H1 + t];
    b += xv * W1[(D_IN + k) * H1 + t];
  }
  Apre[i * H1 + t] = a;
  Bpre[i * H1 + t] = b;
}

// ---------------------------------------------------------------------------
// Straight-through hard Gumbel-softmax over 2 classes, returns y[...,1].
// ---------------------------------------------------------------------------
__device__ __forceinline__ float gumbel_pick(float z0, float z1, float u0, float u1) {
  float g0 = -logf(-logf(u0 + EPSG) + EPSG);
  float g1 = -logf(-logf(u1 + EPSG) + EPSG);
  float t0 = z0 + g0;          // TAU == 1.0
  float t1 = z1 + g1;
  float mx = fmaxf(t0, t1);
  float e0 = expf(t0 - mx);
  float e1 = expf(t1 - mx);
  float inv = 1.0f / (e0 + e1);
  float s0 = e0 * inv, s1 = e1 * inv;
  float hard1 = (s1 > s0) ? 1.0f : 0.0f;
  return (hard1 - s1) + s1;
}

// Layer 1 for one 16-pair m-tile: buf[m][k] = relu(Apre[fixed][k] + Bpre[col+m][k]) as f16
__device__ __forceinline__ void layer1_tile(const float* __restrict__ Apre,
                                            const float* __restrict__ Bpre,
                                            int fixedRow, int colBase, int lane,
                                            _Float16* __restrict__ buf) {
  const f32x4 aV = *(const f32x4*)(Apre + fixedRow * H1 + 4 * lane);
#pragma unroll
  for (int m = 0; m < 16; ++m) {
    const f32x4 bV = *(const f32x4*)(Bpre + (colBase + m) * H1 + 4 * lane);
    union { _Float16 h[4]; i32x2 q; } pk;
    pk.h[0] = (_Float16)fmaxf(aV.x + bV.x, 0.f);
    pk.h[1] = (_Float16)fmaxf(aV.y + bV.y, 0.f);
    pk.h[2] = (_Float16)fmaxf(aV.z + bV.z, 0.f);
    pk.h[3] = (_Float16)fmaxf(aV.w + bV.w, 0.f);
    *(i32x2*)(buf + m * H1S + 4 * lane) = pk.q;  // ds_store_b64, conflict-free
  }
}

// ---------------------------------------------------------------------------
// Kernel 2: per 16x16 pair tile, layers 2-3 via WMMA, symmetrize, gumbel.
// Software-pipelined: layer1(t+1) into the alternate buffer issues under the
// WMMAs of tile t (WMMA co-executes with VALU/DS; DS ops are in-order/wave).
// ---------------------------------------------------------------------------
__global__ __launch_bounds__(256)
void pair_mlp_kernel(const float* __restrict__ Apre,
                     const float* __restrict__ Bpre,
                     const float* __restrict__ W2,
                     const float* __restrict__ b2,
                     const float* __restrict__ W3,
                     const float* __restrict__ b3,
                     const float* __restrict__ u,
                     float* __restrict__ out) {
  const int bi = blockIdx.x;
  const int bj = blockIdx.y;
  if (bi > bj) return;  // uniform across block: safe vs barriers

  __shared__ __align__(16) _Float16 sH1[8][2][16 * H1S]; // per-wave double-buffered staging
  __shared__ __align__(16) _Float16 sW2T[H2 * H1S];      // W2 transposed [n][k], f16
  __shared__ float sW3[H2 * 2];
  __shared__ float sB2[H2];
  __shared__ float sB3[2];
  __shared__ float sLog[2][16][16][2];                   // [T1/T2][fixed][m][ch]

  const int tid = threadIdx.x;

  // Stage constants into LDS.
  for (int e = tid; e < H1 * H2; e += 256) {             // sW2T[n][k] = W2[k][n]
    int k = e >> 6;
    int n = e & 63;
    sW2T[n * H1S + k] = (_Float16)W2[k * H2 + n];
  }
  for (int e = tid; e < H2 * 2; e += 256) sW3[e] = W3[e];
  if (tid < H2) sB2[tid] = b2[tid];
  if (tid < 2)  sB3[tid] = b3[tid];
  __syncthreads();

  const int wave   = tid >> 5;
  const int lane   = tid & 31;
  const int isT2   = wave >> 2;                 // 0: rows from I-block; 1: rows from J-block
  const int wquad  = wave & 3;
  const int rowBase = (isT2 ? bj : bi) * 16;
  const int colBase = (isT2 ? bi : bj) * 16;

  const int a_half = lane >> 4;                 // which 16-lane half
  const int lm     = lane & 15;
  const int p3     = lane >> 1;                 // layer-3 pair index
  const int c3     = lane & 1;                  // layer-3 channel

  // Hoisted per-lane constants.
  float bb[4];
#pragma unroll
  for (int nt = 0; nt < 4; ++nt) bb[nt] = sB2[nt * 16 + lm];
  const float zb3 = sB3[c3];

  // Prologue: layer 1 for tile 0.
  layer1_tile(Apre, Bpre, rowBase + wquad * 4, colBase, lane, sH1[wave][0]);

  for (int t = 0; t < 4; ++t) {
    _Float16* cur = sH1[wave][t & 1];

    // ---- Layer 2: h2(16x64) = relu(h1(16x128) @ W2(128x64) + b2) via WMMA
    v8f acc[4];
#pragma unroll
    for (int nt = 0; nt < 4; ++nt) acc[nt] = (v8f){};
#pragma unroll
    for (int kk = 0; kk < 4; ++kk) {
      // A-frag (16x32 f16): lane 16a+m holds K=8a..8a+7 (v0-3), 16+8a.. (v4-7)
      union { v16h h; i32x4 q[2]; } af;
      const _Float16* ap = cur + lm * H1S + kk * 32 + 8 * a_half;
      af.q[0] = *(const i32x4*)(ap);
      af.q[1] = *(const i32x4*)(ap + 16);
#pragma unroll
      for (int nt = 0; nt < 4; ++nt) {
        // B-frag (32x16 f16): lane 16a+n holds K=8a..8a+7 (v0-3), 16+8a.. (v4-7) of column n
        union { v16h h; i32x4 q[2]; } bf;
        const _Float16* bp = sW2T + (nt * 16 + lm) * H1S + kk * 32 + 8 * a_half;
        bf.q[0] = *(const i32x4*)(bp);
        bf.q[1] = *(const i32x4*)(bp + 16);
        acc[nt] = __builtin_amdgcn_wmma_f32_16x16x32_f16(
            false, af.h, false, bf.h, (short)0, acc[nt], false, false);
      }
    }

    // ---- Software pipeline: layer 1 of next tile into the other buffer.
    // Independent of acc -> issues/co-executes while the WMMAs drain.
    if (t < 3)
      layer1_tile(Apre, Bpre, rowBase + wquad * 4 + t + 1, colBase, lane,
                  sH1[wave][(t + 1) & 1]);

    // ---- bias + relu, scatter h2 (f32) into cur (h1 reads done: DS in-order)
    float* h2 = (float*)cur;
#pragma unroll
    for (int nt = 0; nt < 4; ++nt) {
      const int col = nt * 16 + lm;
#pragma unroll
      for (int v = 0; v < 8; ++v) {
        const int mrow = 8 * a_half + v;
        h2[mrow * H2S + col] = fmaxf(acc[nt][v] + bb[nt], 0.f);
      }
    }

    // ---- Layer 3: logits[p][c] = h2[p] . W3[:,c] + b3[c]  (2 lanes per pair)
    {
      float z = zb3;
#pragma unroll 8
      for (int k = 0; k < H2; ++k)
        z += h2[p3 * H2S + k] * sW3[k * 2 + c3];
      sLog[isT2][wquad * 4 + t][p3][c3] = z;
    }
  }

  __syncthreads();

  // ---- symmetrize + gumbel straight-through, write both (i,j) and (j,i)
  {
    const int a = tid >> 4;
    const int b = tid & 15;
    const float z0 = 0.5f * (sLog[0][a][b][0] + sLog[1][b][a][0]);
    const float z1 = 0.5f * (sLog[0][a][b][1] + sLog[1][b][a][1]);
    const int i = bi * 16 + a;
    const int j = bj * 16 + b;
    const f32x2 uij = *(const f32x2*)(u + (i * N_PTS + j) * 2);
    out[i * N_PTS + j] = gumbel_pick(z0, z1, uij.x, uij.y);
    const f32x2 uji = *(const f32x2*)(u + (j * N_PTS + i) * 2);
    out[j * N_PTS + i] = gumbel_pick(z0, z1, uji.x, uji.y);
  }
}

extern "C" void kernel_launch(void* const* d_in, const int* in_sizes, int n_in,
                              void* d_out, int out_size, void* d_ws, size_t ws_size,
                              hipStream_t stream) {
  const float* X  = (const float*)d_in[0];
  const float* W1 = (const float*)d_in[1];
  const float* b1 = (const float*)d_in[2];
  const float* W2 = (const float*)d_in[3];
  const float* b2 = (const float*)d_in[4];
  const float* W3 = (const float*)d_in[5];
  const float* b3 = (const float*)d_in[6];
  const float* u  = (const float*)d_in[7];
  float* out  = (float*)d_out;
  float* Apre = (float*)d_ws;                 // 1024*128 f32
  float* Bpre = Apre + N_PTS * H1;            // 1024*128 f32  (1 MB total)

  precompute_AB<<<N_PTS, H1, 0, stream>>>(X, W1, b1, Apre, Bpre);

  dim3 grid(N_PTS / 16, N_PTS / 16);
  pair_mlp_kernel<<<grid, 256, 0, stream>>>(Apre, Bpre, W2, b2, W3, b3, u, out);
}
